// GCNNet_79680233275798
// MI455X (gfx1250) — matrix operand order repaired and checked by reading, not compiled, not run on hardware
//
#include <hip/hip_runtime.h>
#include <hip/hip_bf16.h>

typedef __attribute__((ext_vector_type(16))) _Float16 v16h;
typedef __attribute__((ext_vector_type(8)))  _Float16 v8h;
typedef __attribute__((ext_vector_type(8)))  float    v8f;

#define N_NODES  100000
#define F_IN     78
#define KPAD     96      // F_IN padded to multiple of 32
#define F_HID    156
#define NPAD     160     // F_HID padded to multiple of 16
#define KPAD2    160     // F_HID padded to multiple of 32 (FC reduction dim)
#define N_GRAPHS 64
#define EMB      32

// ---------------------------------------------------------------------------
// K0: re-init workspace every call (graph-replay safe): agg=0, deg=1 (self
// loop), pooled=0.
// ---------------------------------------------------------------------------
__global__ void k_init(float* __restrict__ deg, float* __restrict__ agg,
                       float* __restrict__ pooled, long long n_agg) {
    long long i = (long long)blockIdx.x * blockDim.x + threadIdx.x;
    if (i < n_agg) agg[i] = 0.0f;
    if (i < N_NODES) deg[i] = 1.0f;                 // self-loop contribution
    if (i < (long long)N_GRAPHS * F_HID) pooled[i] = 0.0f;
}

// ---------------------------------------------------------------------------
// Prep: x -> f16 zero-padded [N][96]; W -> f16 transposed+padded [160][96].
// Done once per call; converts happen here (bandwidth-trivial) instead of
// inside the WMMA loop.
// ---------------------------------------------------------------------------
__global__ void k_prep_x(const float* __restrict__ x, _Float16* __restrict__ x_h) {
    long long i = (long long)blockIdx.x * blockDim.x + threadIdx.x;
    if (i >= (long long)N_NODES * KPAD) return;
    int m = (int)(i / KPAD), k = (int)(i % KPAD);
    x_h[i] = (k < F_IN) ? (_Float16)x[(size_t)m * F_IN + k] : (_Float16)0.0f;
}

__global__ void k_prep_w(const float* __restrict__ W, _Float16* __restrict__ Wt_h) {
    int i = blockIdx.x * blockDim.x + threadIdx.x;
    if (i >= NPAD * KPAD) return;
    int c = i / KPAD, k = i % KPAD;
    Wt_h[i] = (c < F_HID && k < F_IN) ? (_Float16)W[(size_t)k * F_HID + c]
                                      : (_Float16)0.0f;
}

// ---------------------------------------------------------------------------
// K1: in-degree (+1 via init). K2: deg -> rsqrt(deg) in place.
// ---------------------------------------------------------------------------
__global__ void k_degree(const int* __restrict__ dst, int E, float* __restrict__ deg) {
    int e = blockIdx.x * blockDim.x + threadIdx.x;
    if (e < E) atomicAdd(&deg[dst[e]], 1.0f);
}

__global__ void k_dinv(float* __restrict__ deg) {
    int i = blockIdx.x * blockDim.x + threadIdx.x;
    if (i < N_NODES) deg[i] = rsqrtf(deg[i]);
}

// ---------------------------------------------------------------------------
// K3: h = x @ W via v_wmma_f32_16x16x32_f16. Block = 16 rows x 160 cols,
// 320 threads = 10 waves, wave w owns column tile [16w,16w+16). Operands are
// pre-packed so the loop is 4x b128 load + 1 wmma per k-step, no divergence
// (EXEC stays all-1s around WMMA).
//   A elem j: K = k0 + 8*half + (j<8 ? j : 16 + j-8)  -> two contiguous runs
//   B elem j: K = k0 + 8*half + (j<8 ? j : 16 + j-8)  -> two contiguous runs
// ---------------------------------------------------------------------------
__global__ void __launch_bounds__(320) k_gemm1(const _Float16* __restrict__ x_h,
                                               const _Float16* __restrict__ Wt_h,
                                               float* __restrict__ h) {
    const int lane = threadIdx.x & 31;
    const int wave = threadIdx.x >> 5;      // 0..9 -> N tile
    const int half = lane >> 4;
    const int lr   = lane & 15;
    const int row0 = blockIdx.x * 16;       // 100000 / 16 = 6250 exact
    const int m    = row0 + lr;             // A row for this lane
    const int col  = wave * 16 + lr;        // 0..159 (padded; store guarded)

    const _Float16* ap = x_h  + (size_t)m   * KPAD + 8  * half;
    const _Float16* bp = Wt_h + (size_t)col * KPAD + 8  * half;

    v8f c = {};
    #pragma unroll
    for (int k0 = 0; k0 < KPAD; k0 += 32) {
        v8h alo = *(const v8h*)(ap + k0);
        v8h ahi = *(const v8h*)(ap + k0 + 16);
        v8h blo = *(const v8h*)(bp + k0);
        v8h bhi = *(const v8h*)(bp + k0 + 16);
        v16h a = __builtin_shufflevector(alo, ahi, 0,1,2,3,4,5,6,7,
                                         8,9,10,11,12,13,14,15);
        v16h b = __builtin_shufflevector(blo, bhi, 0,1,2,3,4,5,6,7,
                                         8,9,10,11,12,13,14,15);
        c = __builtin_amdgcn_wmma_f32_16x16x32_f16(false, a, false, b,
                                                   (short)0, c, false, false);
    }
    if (col < F_HID) {
        #pragma unroll
        for (int v = 0; v < 8; ++v)
            h[(size_t)(row0 + v + 8 * half) * F_HID + col] = c[v];
    }
}

// ---------------------------------------------------------------------------
// K4: edge scatter. One wave per edge; lanes stride the 156 features.
// Coalesced row gather + coalesced global_atomic_add_f32. h/agg (62 MB each)
// are L2-resident (192 MB L2) -> this phase rides L2 bandwidth.
// ---------------------------------------------------------------------------
__global__ void __launch_bounds__(256) k_scatter(const int* __restrict__ ei, int E,
                                                 const float* __restrict__ dinv,
                                                 const float* __restrict__ h,
                                                 float* __restrict__ agg) {
    int wid  = (int)((blockIdx.x * (unsigned)blockDim.x + threadIdx.x) >> 5);
    int lane = threadIdx.x & 31;
    if (wid >= E) return;
    int s = ei[wid];
    int d = ei[E + wid];
    float nrm = dinv[s] * dinv[d];
    const float* hrow = h + (size_t)s * F_HID;
    float*       arow = agg + (size_t)d * F_HID;
    #pragma unroll
    for (int f = lane; f < F_HID; f += 32)
        atomicAdd(&arow[f], hrow[f] * nrm);
}

// ---------------------------------------------------------------------------
// K5: self-loop + bias + relu + segment max-pool. One wave per node.
// relu output >= 0 and pooled init 0.0 => uint atomicMax on the bit pattern
// is order-preserving and reproduces the reference's empty-segment guard.
// ---------------------------------------------------------------------------
__global__ void __launch_bounds__(256) k_pool(const float* __restrict__ agg,
                                              const float* __restrict__ h,
                                              const float* __restrict__ dinv,
                                              const float* __restrict__ bias,
                                              const int* __restrict__ batch,
                                              float* __restrict__ pooled) {
    int wid  = (int)((blockIdx.x * (unsigned)blockDim.x + threadIdx.x) >> 5);
    int lane = threadIdx.x & 31;
    if (wid >= N_NODES) return;
    float di   = dinv[wid];
    float self = di * di;                   // self-loop norm
    int   g    = batch[wid];
    unsigned*    prow = (unsigned*)(pooled + (size_t)g * F_HID);
    const float* arow = agg + (size_t)wid * F_HID;
    const float* hrow = h   + (size_t)wid * F_HID;
    #pragma unroll
    for (int f = lane; f < F_HID; f += 32) {
        float v = fmaxf(arow[f] + hrow[f] * self + bias[f], 0.0f);
        atomicMax(&prow[f], __float_as_uint(v));
    }
}

// ---------------------------------------------------------------------------
// Prep for FC: pooled -> f16 padded [64][160]; fc_W -> f16 transposed [32][160].
// ---------------------------------------------------------------------------
__global__ void k_prep_fc(const float* __restrict__ pooled,
                          const float* __restrict__ fcW,
                          _Float16* __restrict__ pooled_h,
                          _Float16* __restrict__ fcWt_h) {
    int i = blockIdx.x * blockDim.x + threadIdx.x;
    if (i < N_GRAPHS * KPAD2) {
        int g = i / KPAD2, k = i % KPAD2;
        pooled_h[i] = (k < F_HID) ? (_Float16)pooled[(size_t)g * F_HID + k]
                                  : (_Float16)0.0f;
    } else if (i < N_GRAPHS * KPAD2 + EMB * KPAD2) {
        int j = i - N_GRAPHS * KPAD2;
        int c = j / KPAD2, k = j % KPAD2;
        fcWt_h[j] = (k < F_HID) ? (_Float16)fcW[(size_t)k * EMB + c]
                                : (_Float16)0.0f;
    }
}

// ---------------------------------------------------------------------------
// K6: out = relu(pooled @ fc_W + fc_b), 64x156x32 via WMMA. One block,
// 8 waves = 4 M-tiles x 2 N-tiles. K loop: 5 steps of 32 over padded 160.
// ---------------------------------------------------------------------------
__global__ void __launch_bounds__(256) k_fc(const _Float16* __restrict__ pooled_h,
                                            const _Float16* __restrict__ fcWt_h,
                                            const float* __restrict__ fcb,
                                            float* __restrict__ out) {
    const int lane  = threadIdx.x & 31;
    const int wave  = threadIdx.x >> 5;     // 0..7
    const int mtile = wave >> 1;            // 0..3
    const int ntile = wave & 1;             // 0..1
    const int half  = lane >> 4;
    const int lr    = lane & 15;
    const int m     = mtile * 16 + lr;
    const int col   = ntile * 16 + lr;      // < 32 always

    const _Float16* ap = pooled_h + (size_t)m   * KPAD2 + 8  * half;
    const _Float16* bp = fcWt_h   + (size_t)col * KPAD2 + 8  * half;

    v8f c = {};
    #pragma unroll
    for (int k0 = 0; k0 < KPAD2; k0 += 32) {
        v8h alo = *(const v8h*)(ap + k0);
        v8h ahi = *(const v8h*)(ap + k0 + 16);
        v8h blo = *(const v8h*)(bp + k0);
        v8h bhi = *(const v8h*)(bp + k0 + 16);
        v16h a = __builtin_shufflevector(alo, ahi, 0,1,2,3,4,5,6,7,
                                         8,9,10,11,12,13,14,15);
        v16h b = __builtin_shufflevector(blo, bhi, 0,1,2,3,4,5,6,7,
                                         8,9,10,11,12,13,14,15);
        c = __builtin_amdgcn_wmma_f32_16x16x32_f16(false, a, false, b,
                                                   (short)0, c, false, false);
    }
    #pragma unroll
    for (int v = 0; v < 8; ++v) {
        int mr = mtile * 16 + v + 8 * half;
        out[mr * EMB + col] = fmaxf(c[v] + fcb[col], 0.0f);
    }
}

// ---------------------------------------------------------------------------
extern "C" void kernel_launch(void* const* d_in, const int* in_sizes, int n_in,
                              void* d_out, int out_size, void* d_ws, size_t ws_size,
                              hipStream_t stream) {
    const float* x    = (const float*)d_in[0];
    const int*   ei   = (const int*)d_in[1];     // [2, E] flat: src row, dst row
    const int*   batch= (const int*)d_in[2];
    const float* W    = (const float*)d_in[3];
    const float* bias = (const float*)d_in[4];
    const float* fcW  = (const float*)d_in[5];
    const float* fcb  = (const float*)d_in[6];
    float*       out  = (float*)d_out;

    const int E = in_sizes[1] / 2;               // 1,600,000

    // workspace layout
    float* ws     = (float*)d_ws;
    float* deg    = ws;                                    // N f32 (becomes dinv)
    float* h      = ws + 100352;                           // N*156 f32
    float* agg    = h  + (size_t)N_NODES * F_HID;          // N*156 f32
    float* pooled = agg + (size_t)N_NODES * F_HID;         // 64*156 f32
    _Float16* x_h      = (_Float16*)(pooled + 10240);      // N*96 f16
    _Float16* Wt_h     = x_h  + (size_t)N_NODES * KPAD;    // 160*96 f16
    _Float16* pooled_h = Wt_h + NPAD * KPAD;               // 64*160 f16
    _Float16* fcWt_h   = pooled_h + N_GRAPHS * KPAD2;      // 32*160 f16

    const long long n_agg = (long long)N_NODES * F_HID;    // 15.6M

    k_init<<<(int)((n_agg + 255) / 256), 256, 0, stream>>>(deg, agg, pooled, n_agg);
    k_prep_x<<<(int)(((long long)N_NODES * KPAD + 255) / 256), 256, 0, stream>>>(x, x_h);
    k_prep_w<<<(NPAD * KPAD + 255) / 256, 256, 0, stream>>>(W, Wt_h);
    k_degree<<<(E + 255) / 256, 256, 0, stream>>>(ei + E, E, deg);
    k_dinv<<<(N_NODES + 255) / 256, 256, 0, stream>>>(deg);
    k_gemm1<<<N_NODES / 16, 320, 0, stream>>>(x_h, Wt_h, h);
    k_scatter<<<(E * 32 + 255) / 256, 256, 0, stream>>>(ei, E, deg, h, agg);
    k_pool<<<(N_NODES * 32 + 255) / 256, 256, 0, stream>>>(agg, h, deg, bias, batch, pooled);
    k_prep_fc<<<((N_GRAPHS + EMB) * KPAD2 + 255) / 256, 256, 0, stream>>>(pooled, fcW,
                                                                          pooled_h, fcWt_h);
    k_fc<<<1, 256, 0, stream>>>(pooled_h, fcWt_h, fcb, out);
}
